// GATFraudDetector_26096221290643
// MI455X (gfx1250) — matrix operand order repaired and checked by reading, not compiled, not run on hardware
//
#include <hip/hip_runtime.h>
#include <hip/hip_bf16.h>

#define N_NODES 50000
#define N_EDGES 800000
#define E_TOT   850000   /* edges + self loops */
#define IN_DIM  128
#define HID     64
#define HEADS   4
#define C1      256      /* HEADS*HID */
#define NEG_SLOPE 0.2f
#define EPS_A 1e-16f

typedef __attribute__((ext_vector_type(16))) _Float16 v16h;
typedef __attribute__((ext_vector_type(8)))  float    v8f;
typedef __attribute__((ext_vector_type(4)))  int      v4i;

// pointer-to-v4i in global (AS1) / LDS (AS3) address spaces
typedef __attribute__((address_space(1))) v4i* g_v4i_p;
typedef __attribute__((address_space(3))) v4i* l_v4i_p;

#if defined(__has_builtin)
#  if __has_builtin(__builtin_amdgcn_global_load_async_to_lds_b128)
#    define HAVE_ASYNC_LDS 1
#  endif
#endif
#ifndef HAVE_ASYNC_LDS
#  define HAVE_ASYNC_LDS 0
#endif

// ---------------------------------------------------------------------------
// helpers
// ---------------------------------------------------------------------------
__device__ __forceinline__ void atomic_max_f32(float* addr, float v) {
  // signed-int max for v>=0, unsigned-int min for v<0; buffer must be
  // initialized to -inf (0xFF800000). Works for mixed signs.
  if (v >= 0.0f) atomicMax((int*)addr, __float_as_int(v));
  else           atomicMin((unsigned int*)addr, __float_as_uint(v));
}

__device__ __forceinline__ void edge_sd(const int* __restrict__ ei, int e,
                                        int& s, int& d) {
  if (e < N_EDGES) { s = ei[e]; d = ei[N_EDGES + e]; }
  else             { s = e - N_EDGES; d = s; }          // self loop
}

// ---------------------------------------------------------------------------
// init / prep
// ---------------------------------------------------------------------------
__global__ void fill_u32(unsigned int* p, unsigned int val, int n) {
  int i = blockIdx.x * blockDim.x + threadIdx.x;
  if (i < n) p[i] = val;
}

// W1T[n][k] = (f16) W1[k][n]   (256 x 128)
__global__ void prep_w1t(const float* __restrict__ W1, _Float16* __restrict__ W1T) {
  int t = blockIdx.x * blockDim.x + threadIdx.x;   // 32768 threads
  if (t >= C1 * IN_DIM) return;
  int n = t >> 7, k = t & 127;
  W1T[t] = (_Float16)W1[k * C1 + n];
}

// ---------------------------------------------------------------------------
// Layer-1 GEMM:  H[50000][256] = x[50000][128] @ W1 + b1   (WMMA f16->f32)
// one 16x16 output tile per wave; 8 waves/block cover 8 N-tiles.
// B panel (128 cols x 128 k, f16 = 32KB) staged into LDS, via the CDNA5
// async global->LDS path (ASYNCcnt) when the toolchain exposes it.
// ---------------------------------------------------------------------------
__global__ __launch_bounds__(256) void gemm1_wmma(
    const float* __restrict__ x, const _Float16* __restrict__ W1T,
    const float* __restrict__ b1, float* __restrict__ H) {
  __shared__ _Float16 Bsh[128 * IN_DIM];        // 32 KB per block

  int tid   = threadIdx.x;
  int lane  = threadIdx.x & 31;
  int wv    = threadIdx.x >> 5;
  int mtile = blockIdx.x >> 1;                  // 3125 M tiles
  int nhalf = blockIdx.x & 1;                   // which 8 N-tiles
  int hi    = lane >> 4;                        // half-wave selector
  int mn    = lane & 15;

  // ---- stage B panel: 128 rows of W1T (cols nhalf*128 ..), 2048 x 16B ----
  const uint4* gsrc = (const uint4*)(W1T + (size_t)(nhalf * 128) * IN_DIM);
#if HAVE_ASYNC_LDS
#pragma unroll
  for (int it = 0; it < 8; ++it) {
    int chunk = it * 256 + tid;
    __builtin_amdgcn_global_load_async_to_lds_b128(
        (g_v4i_p)(gsrc + chunk),
        (l_v4i_p)(Bsh + (size_t)chunk * 8),
        0, 0);
  }
#  if __has_builtin(__builtin_amdgcn_s_wait_asynccnt)
  __builtin_amdgcn_s_wait_asynccnt(0);
#  else
  asm volatile("s_wait_asynccnt 0" ::: "memory");
#  endif
#else
#pragma unroll
  for (int it = 0; it < 8; ++it) {
    int chunk = it * 256 + tid;
    ((uint4*)Bsh)[chunk] = gsrc[chunk];
  }
#endif
  __syncthreads();

  const float*    xrow = x   + (size_t)(mtile * 16 + mn) * IN_DIM;
  const _Float16* wrow = Bsh + (size_t)(wv * 16 + mn) * IN_DIM + hi * 16;

  v8f acc = {};
#pragma unroll
  for (int ks = 0; ks < IN_DIM; ks += 32) {
    // A fragment: 16 consecutive K {kb..kb+15}, matching B
    int kb = ks + hi * 16;
    v16h A;
#pragma unroll
    for (int j = 0; j < 8; ++j) {
      A[j]     = (_Float16)xrow[kb + j];
      A[j + 8] = (_Float16)xrow[kb + 8 + j];
    }
    // B fragment: 16 consecutive K for this lane's column (32B aligned, LDS)
    v16h B = *(const v16h*)(wrow + ks);
    acc = __builtin_amdgcn_wmma_f32_16x16x32_f16(
        false, A, false, B, (short)0, acc, false, false);
  }

  int   col  = (nhalf * 8 + wv) * 16 + mn;
  float bias = b1[col];
#pragma unroll
  for (int r = 0; r < 8; ++r) {                 // C/D layout: VGPR r -> M=r(+8)
    int row = mtile * 16 + hi * 8 + r;
    H[(size_t)row * C1 + col] = acc[r] + bias;
  }
}

// ---------------------------------------------------------------------------
// node attention logits layer 1: as1/ad1[n][h] = <H[n][h], a_src/a_dst[h]>
// one wave per node, coalesced column strides, shuffle reduction
// ---------------------------------------------------------------------------
__global__ __launch_bounds__(256) void alpha1_kernel(
    const float* __restrict__ H,
    const float* __restrict__ a_src, const float* __restrict__ a_dst,
    float* __restrict__ as1, float* __restrict__ ad1) {
  int n    = (blockIdx.x * blockDim.x + threadIdx.x) >> 5;
  int lane = threadIdx.x & 31;
  if (n >= N_NODES) return;
  const float* row = H + (size_t)n * C1;
  float ps[HEADS] = {0.f, 0.f, 0.f, 0.f};
  float pd[HEADS] = {0.f, 0.f, 0.f, 0.f};
#pragma unroll
  for (int j = 0; j < 8; ++j) {                 // head index uniform per pass
    int h  = j >> 1;
    int k  = (j & 1) * 32 + lane;               // index within head
    float hv = row[j * 32 + lane];
    ps[h] += hv * a_src[h * HID + k];
    pd[h] += hv * a_dst[h * HID + k];
  }
#pragma unroll
  for (int h = 0; h < HEADS; ++h) {
#pragma unroll
    for (int off = 16; off > 0; off >>= 1) {
      ps[h] += __shfl_xor(ps[h], off, 32);
      pd[h] += __shfl_xor(pd[h], off, 32);
    }
  }
  if (lane == 0) {
#pragma unroll
    for (int h = 0; h < HEADS; ++h) {
      as1[n * HEADS + h] = ps[h];
      ad1[n * HEADS + h] = pd[h];
    }
  }
}

// ---------------------------------------------------------------------------
// layer-1 segment softmax (max, then exp-sum), one thread per edge
// ---------------------------------------------------------------------------
__global__ void edge_max1(const int* __restrict__ ei,
                          const float* __restrict__ as1, const float* __restrict__ ad1,
                          float* __restrict__ m1) {
  int e = blockIdx.x * blockDim.x + threadIdx.x;
  if (e >= E_TOT) return;
  int s, d; edge_sd(ei, e, s, d);
  float4 a = *(const float4*)(as1 + s * 4);
  float4 b = *(const float4*)(ad1 + d * 4);
  float v[4] = {a.x + b.x, a.y + b.y, a.z + b.z, a.w + b.w};
#pragma unroll
  for (int h = 0; h < 4; ++h) {
    float vv = v[h] > 0.f ? v[h] : NEG_SLOPE * v[h];
    atomic_max_f32(&m1[d * 4 + h], vv);
  }
}

__global__ void edge_sum1(const int* __restrict__ ei,
                          const float* __restrict__ as1, const float* __restrict__ ad1,
                          const float* __restrict__ m1, float* __restrict__ den1) {
  int e = blockIdx.x * blockDim.x + threadIdx.x;
  if (e >= E_TOT) return;
  int s, d; edge_sd(ei, e, s, d);
  float4 a  = *(const float4*)(as1 + s * 4);
  float4 b  = *(const float4*)(ad1 + d * 4);
  float4 mm = *(const float4*)(m1 + d * 4);
  float v[4]  = {a.x + b.x, a.y + b.y, a.z + b.z, a.w + b.w};
  float mx[4] = {mm.x, mm.y, mm.z, mm.w};
#pragma unroll
  for (int h = 0; h < 4; ++h) {
    float vv = v[h] > 0.f ? v[h] : NEG_SLOPE * v[h];
    atomicAdd(&den1[d * 4 + h], __expf(vv - mx[h]));
  }
}

// ---------------------------------------------------------------------------
// layer-1 aggregation: one wave per edge; 8 strided passes over 256 columns
// ---------------------------------------------------------------------------
__global__ __launch_bounds__(256) void edge_agg1(
    const int* __restrict__ ei,
    const float* __restrict__ as1, const float* __restrict__ ad1,
    const float* __restrict__ m1,  const float* __restrict__ den1,
    const float* __restrict__ H,   float* __restrict__ out1) {
  int wid  = (blockIdx.x * blockDim.x + threadIdx.x) >> 5;
  int lane = threadIdx.x & 31;
  if (wid >= E_TOT) return;
  int s, d; edge_sd(ei, wid, s, d);

  float al = 0.f;
  if (lane < HEADS) {
    float v = as1[s * 4 + lane] + ad1[d * 4 + lane];
    v = v > 0.f ? v : NEG_SLOPE * v;
    float ex = __expf(v - m1[d * 4 + lane]);
    al = ex / (den1[d * 4 + lane] + EPS_A);
  }
  float alphas[4];
  alphas[0] = __shfl(al, 0, 32);
  alphas[1] = __shfl(al, 1, 32);
  alphas[2] = __shfl(al, 2, 32);
  alphas[3] = __shfl(al, 3, 32);

  const float* hs = H    + (size_t)s * C1;
  float*       od = out1 + (size_t)d * C1;
#pragma unroll
  for (int j = 0; j < 8; ++j) {             // head index uniform per pass
    int col = j * 32 + lane;
    atomicAdd(&od[col], hs[col] * alphas[j >> 1]);
  }
}

// ---------------------------------------------------------------------------
// layer-2 projection: h2[n] = <relu(out1[n] + b1), W2>; as2/ad2 = h2 * a_*2
// one wave per node, shuffle reduction
// ---------------------------------------------------------------------------
__global__ __launch_bounds__(256) void node2_kernel(
    const float* __restrict__ out1, const float* __restrict__ b1,
    const float* __restrict__ W2,
    const float* __restrict__ a_src2, const float* __restrict__ a_dst2,
    float* __restrict__ h2, float* __restrict__ as2, float* __restrict__ ad2) {
  int n    = (blockIdx.x * blockDim.x + threadIdx.x) >> 5;
  int lane = threadIdx.x & 31;
  if (n >= N_NODES) return;
  const float* row = out1 + (size_t)n * C1;
  float p = 0.f;
#pragma unroll
  for (int j = 0; j < 8; ++j) {
    int c = j * 32 + lane;
    float y = row[c] + b1[c];
    y = y > 0.f ? y : 0.f;                  // ReLU between layers
    p += y * W2[c];
  }
#pragma unroll
  for (int off = 16; off > 0; off >>= 1) p += __shfl_xor(p, off, 32);
  if (lane == 0) { h2[n] = p; as2[n] = p * a_src2[0]; ad2[n] = p * a_dst2[0]; }
}

// ---------------------------------------------------------------------------
// layer-2 softmax + aggregation (scalar per edge)
// ---------------------------------------------------------------------------
__global__ void edge_max2(const int* __restrict__ ei,
                          const float* __restrict__ as2, const float* __restrict__ ad2,
                          float* __restrict__ m2) {
  int e = blockIdx.x * blockDim.x + threadIdx.x;
  if (e >= E_TOT) return;
  int s, d; edge_sd(ei, e, s, d);
  float v = as2[s] + ad2[d];
  v = v > 0.f ? v : NEG_SLOPE * v;
  atomic_max_f32(&m2[d], v);
}

__global__ void edge_sum2(const int* __restrict__ ei,
                          const float* __restrict__ as2, const float* __restrict__ ad2,
                          const float* __restrict__ m2, float* __restrict__ den2) {
  int e = blockIdx.x * blockDim.x + threadIdx.x;
  if (e >= E_TOT) return;
  int s, d; edge_sd(ei, e, s, d);
  float v = as2[s] + ad2[d];
  v = v > 0.f ? v : NEG_SLOPE * v;
  atomicAdd(&den2[d], __expf(v - m2[d]));
}

__global__ void edge_agg2(const int* __restrict__ ei,
                          const float* __restrict__ as2, const float* __restrict__ ad2,
                          const float* __restrict__ m2, const float* __restrict__ den2,
                          const float* __restrict__ h2, float* __restrict__ out2) {
  int e = blockIdx.x * blockDim.x + threadIdx.x;
  if (e >= E_TOT) return;
  int s, d; edge_sd(ei, e, s, d);
  float v = as2[s] + ad2[d];
  v = v > 0.f ? v : NEG_SLOPE * v;
  float alpha = __expf(v - m2[d]) / (den2[d] + EPS_A);
  atomicAdd(&out2[d], h2[s] * alpha);
}

__global__ void final_kernel(const float* __restrict__ out2,
                             const float* __restrict__ b2,
                             float* __restrict__ out) {
  int n = blockIdx.x * blockDim.x + threadIdx.x;
  if (n >= N_NODES) return;
  float z = out2[n] + b2[0];
  out[n] = 1.0f / (1.0f + __expf(-z));
}

// ---------------------------------------------------------------------------
// host
// ---------------------------------------------------------------------------
static inline int cdiv(long long a, int b) { return (int)((a + b - 1) / b); }

extern "C" void kernel_launch(void* const* d_in, const int* in_sizes, int n_in,
                              void* d_out, int out_size, void* d_ws, size_t ws_size,
                              hipStream_t stream) {
  const float* x      = (const float*)d_in[0];
  const int*   ei     = (const int*)  d_in[1];
  const float* W1     = (const float*)d_in[2];
  const float* a_src1 = (const float*)d_in[3];
  const float* a_dst1 = (const float*)d_in[4];
  const float* b1     = (const float*)d_in[5];
  const float* W2     = (const float*)d_in[6];
  const float* a_src2 = (const float*)d_in[7];
  const float* a_dst2 = (const float*)d_in[8];
  const float* b2     = (const float*)d_in[9];
  float* out = (float*)d_out;

  // bump allocator over workspace (~107 MB total)
  char*  base = (char*)d_ws;
  size_t off  = 0;
  auto alloc = [&](size_t bytes) -> char* {
    char* p = base + off;
    off = (off + bytes + 255) & ~(size_t)255;
    return p;
  };
  _Float16* W1T = (_Float16*)alloc((size_t)C1 * IN_DIM * sizeof(_Float16));
  float* H    = (float*)alloc((size_t)N_NODES * C1 * sizeof(float));
  float* out1 = (float*)alloc((size_t)N_NODES * C1 * sizeof(float));
  float* as1  = (float*)alloc((size_t)N_NODES * HEADS * sizeof(float));
  float* ad1  = (float*)alloc((size_t)N_NODES * HEADS * sizeof(float));
  float* m1   = (float*)alloc((size_t)N_NODES * HEADS * sizeof(float));
  float* den1 = (float*)alloc((size_t)N_NODES * HEADS * sizeof(float));
  float* h2   = (float*)alloc((size_t)N_NODES * sizeof(float));
  float* as2  = (float*)alloc((size_t)N_NODES * sizeof(float));
  float* ad2  = (float*)alloc((size_t)N_NODES * sizeof(float));
  float* m2   = (float*)alloc((size_t)N_NODES * sizeof(float));
  float* den2 = (float*)alloc((size_t)N_NODES * sizeof(float));
  float* out2 = (float*)alloc((size_t)N_NODES * sizeof(float));

  const unsigned NEG_INF = 0xFF800000u;

  // prep + init (all deterministic, re-done every call)
  prep_w1t<<<cdiv(C1 * IN_DIM, 256), 256, 0, stream>>>(W1, W1T);
  fill_u32<<<cdiv(N_NODES * HEADS, 256), 256, 0, stream>>>((unsigned*)m1,   NEG_INF, N_NODES * HEADS);
  fill_u32<<<cdiv(N_NODES * HEADS, 256), 256, 0, stream>>>((unsigned*)den1, 0u,      N_NODES * HEADS);
  fill_u32<<<cdiv((long long)N_NODES * C1, 256), 256, 0, stream>>>((unsigned*)out1, 0u, N_NODES * C1);
  fill_u32<<<cdiv(N_NODES, 256), 256, 0, stream>>>((unsigned*)m2,   NEG_INF, N_NODES);
  fill_u32<<<cdiv(N_NODES, 256), 256, 0, stream>>>((unsigned*)den2, 0u,      N_NODES);
  fill_u32<<<cdiv(N_NODES, 256), 256, 0, stream>>>((unsigned*)out2, 0u,      N_NODES);

  // layer 1
  gemm1_wmma<<<(N_NODES / 16) * 2, 256, 0, stream>>>(x, W1T, b1, H);
  alpha1_kernel<<<cdiv((long long)N_NODES * 32, 256), 256, 0, stream>>>(H, a_src1, a_dst1, as1, ad1);
  edge_max1<<<cdiv(E_TOT, 256), 256, 0, stream>>>(ei, as1, ad1, m1);
  edge_sum1<<<cdiv(E_TOT, 256), 256, 0, stream>>>(ei, as1, ad1, m1, den1);
  edge_agg1<<<cdiv((long long)E_TOT * 32, 256), 256, 0, stream>>>(ei, as1, ad1, m1, den1, H, out1);

  // layer 2
  node2_kernel<<<cdiv((long long)N_NODES * 32, 256), 256, 0, stream>>>(
      out1, b1, W2, a_src2, a_dst2, h2, as2, ad2);
  edge_max2<<<cdiv(E_TOT, 256), 256, 0, stream>>>(ei, as2, ad2, m2);
  edge_sum2<<<cdiv(E_TOT, 256), 256, 0, stream>>>(ei, as2, ad2, m2, den2);
  edge_agg2<<<cdiv(E_TOT, 256), 256, 0, stream>>>(ei, as2, ad2, m2, den2, h2, out2);

  final_kernel<<<cdiv(N_NODES, 256), 256, 0, stream>>>(out2, b2, out);
}
